// SubMAttention3dv2_2972117369408
// MI455X (gfx1250) — matrix operand order repaired and checked by reading, not compiled, not run on hardware
//
#include <hip/hip_runtime.h>
#include <hip/hip_bf16.h>

typedef __bf16 bf16_t;
typedef __attribute__((ext_vector_type(16))) __bf16 v16bf;
typedef __attribute__((ext_vector_type(8)))  __bf16 v8bf;
typedef __attribute__((ext_vector_type(2)))  __bf16 v2bf;
typedef __attribute__((ext_vector_type(8)))  float  v8f;

#define NPTS 60000
#define CCH  128
#define KNBR 32
#define FFD  512
#define COUT 128
#define NH   8
#define DHD  16

// ---- WMMA fragment loaders (CDNA5 ISA 7.12.2, wave32) ----
// A 16x32 bf16: lane holds row m=lane&15; k-pairs at ((p<4)?0:16) + ((lane>>4)*8) + 2*(p&3)
__device__ __forceinline__ v16bf load_afrag(const bf16_t* __restrict__ rowPtr, int lane, int k0) {
  int kb = (lane >> 4) << 3;
  v16bf a;
#pragma unroll
  for (int p = 0; p < 8; ++p) {
    int k = ((p < 4) ? 0 : 16) + kb + ((p & 3) << 1);
    v2bf t = *(const v2bf*)(rowPtr + k0 + k);      // 32-bit load (merges to b128)
    a[2 * p] = t[0];
    a[2 * p + 1] = t[1];
  }
  return a;
}
// B 32x16 bf16: lane holds col n=lane&15; k = (lane>>4)*8 + e (e<8), 16 + (lane>>4)*8 + (e-8) (e>=8) -> 2 x 128-bit loads
__device__ __forceinline__ v16bf load_bfrag(const bf16_t* __restrict__ rowPtr, int lane, int k0) {
  int kb = (lane >> 4) << 3;
  const v8bf* p = (const v8bf*)(rowPtr + k0 + kb);
  v8bf lo = p[0], hi = p[2];
  v16bf b;
#pragma unroll
  for (int e = 0; e < 8; ++e) { b[e] = lo[e]; b[8 + e] = hi[e]; }
  return b;
}
// C/D f32 16x16: row m = (lane>>4)*8 + i, col n = lane&15

// ---------------- specialized GEMM: out[NPTS,128] = A[NPTS,128] @ W[128,128]^T + bias (+resid) ----------------
template <bool HAS_RESID>
__global__ void gemm128_k(const bf16_t* __restrict__ A, const bf16_t* __restrict__ W,
                          const float* __restrict__ bias, const float* __restrict__ resid,
                          float* __restrict__ out) {
  int lane = threadIdx.x & 31;
  int wave = threadIdx.x >> 5;
  int r0 = blockIdx.x * 16;
  int c0 = (blockIdx.y << 6) + wave * 16;
  const bf16_t* arow = A + (size_t)(r0 + (lane & 15)) * CCH;
  const bf16_t* brow = W + (size_t)(c0 + (lane & 15)) * CCH;
  v8f acc = {};
#pragma unroll
  for (int k0 = 0; k0 < CCH; k0 += 32) {
    v16bf a = load_afrag(arow, lane, k0);
    v16bf b = load_bfrag(brow, lane, k0);
    acc = __builtin_amdgcn_wmma_f32_16x16x32_bf16(false, a, false, b, (short)0, acc, false, false);
  }
  int col = c0 + (lane & 15);
  int mBase = r0 + ((lane >> 4) << 3);
  float bs = bias[col];
  float rv[8];
  if (HAS_RESID) {
#pragma unroll
    for (int i = 0; i < 8; ++i) rv[i] = resid[(size_t)(mBase + i) * CCH + col];
  }
#pragma unroll
  for (int i = 0; i < 8; ++i) {
    float v = acc[i] + bs;
    if (HAS_RESID) v += rv[i];
    out[(size_t)(mBase + i) * CCH + col] = v;
  }
}

// ---------------- f32 -> bf16 weight conversion ----------------
__global__ void cvt_bf16_k(const float* __restrict__ src, bf16_t* __restrict__ dst, int n) {
  int i = blockIdx.x * blockDim.x + threadIdx.x;
  if (i < n) dst[i] = (bf16_t)src[i];
}

// ---------------- coords + q = features + relu(coords @ qposW^T + b), q stored bf16 ----------------
__global__ void coords_q_k(const float* __restrict__ features, const int* __restrict__ indices,
                           const float* __restrict__ qpw, const float* __restrict__ qpb,
                           float* __restrict__ coords, bf16_t* __restrict__ qbf) {
  int idx = blockIdx.x * blockDim.x + threadIdx.x;
  if (idx >= NPTS * CCH) return;
  int n = idx >> 7, c = idx & 127;
  const float vox[3] = {0.1f, 0.1f, 0.2f};
  const float pmn[3] = {-40.f, -40.f, -3.f};
  float cd[3];
#pragma unroll
  for (int j = 0; j < 3; ++j)
    cd[j] = ((float)indices[n * 4 + (3 - j)] + 0.5f) * vox[j] + pmn[j];
  if (c < 3) coords[n * 3 + c] = cd[c];
  float p = cd[0] * qpw[c * 3 + 0] + cd[1] * qpw[c * 3 + 1] + cd[2] * qpw[c * 3 + 2] + qpb[c];
  qbf[idx] = (bf16_t)(features[idx] + fmaxf(p, 0.f));
}

// ---------------- per-query attention: gather+pos, WMMA K/V proj, scores, softmax, AV ----------------
__global__ void attn_k(const float* __restrict__ features, const float* __restrict__ coords,
                       const int* __restrict__ key_indices,
                       const float* __restrict__ kpw, const float* __restrict__ kpb,
                       const bf16_t* __restrict__ wkbf, const bf16_t* __restrict__ wvbf,
                       const float* __restrict__ in_proj_b,
                       const float* __restrict__ qh,
                       bf16_t* __restrict__ attOut) {
  __shared__ __align__(16) bf16_t kf[KNBR][CCH];
  __shared__ float  khv[2][KNBR][CCH];
  __shared__ float  rel[KNBR][3];
  __shared__ int    msk[KNBR];
  __shared__ float  sc[NH][KNBR];
  __shared__ float  qrow[CCH];
  int n = blockIdx.x;
  int tid = threadIdx.x, lane = tid & 31, wave = tid >> 5;

  if (tid < KNBR * 3) {
    int k = tid / 3, j = tid % 3;
    int g = key_indices[n * KNBR + k];
    if (j == 0) msk[k] = (g < 0) ? 1 : 0;
    int gg = g < 0 ? 0 : g;
    rel[k][j] = coords[gg * 3 + j] - coords[n * 3 + j];
  }
  if (tid < CCH) qrow[tid] = qh[(size_t)n * CCH + tid];
  __syncthreads();
  for (int idx = tid; idx < KNBR * CCH; idx += blockDim.x) {
    int k = idx >> 7, c = idx & 127;
    int g = key_indices[n * KNBR + k];
    int gg = g < 0 ? 0 : g;
    float f = features[(size_t)gg * CCH + c];
    float p = rel[k][0] * kpw[c * 3 + 0] + rel[k][1] * kpw[c * 3 + 1] +
              rel[k][2] * kpw[c * 3 + 2] + kpb[c];
    kf[k][c] = (bf16_t)(f + fmaxf(p, 0.f));
  }
  __syncthreads();
  // K/V projections: 32 tiles (proj x 2 Mtiles x 8 Ntiles), 8 per wave, K=128
#pragma unroll
  for (int t = 0; t < 8; ++t) {
    int g = wave * 8 + t;
    int proj = g >> 4;
    int rem = g & 15;
    int mi = rem >> 3, ni = rem & 7;
    const bf16_t* W = proj ? wvbf : wkbf;
    const float* bias = in_proj_b + (proj ? 2 * CCH : CCH);
    const bf16_t* arow = &kf[mi * 16 + (lane & 15)][0];
    const bf16_t* brow = W + (size_t)(ni * 16 + (lane & 15)) * CCH;
    v8f acc = {};
#pragma unroll
    for (int k0 = 0; k0 < CCH; k0 += 32) {
      v16bf a = load_afrag(arow, lane, k0);
      v16bf b = load_bfrag(brow, lane, k0);
      acc = __builtin_amdgcn_wmma_f32_16x16x32_bf16(false, a, false, b, (short)0, acc, false, false);
    }
    int col = ni * 16 + (lane & 15);
#pragma unroll
    for (int i = 0; i < 8; ++i) {
      int m = mi * 16 + ((lane >> 4) << 3) + i;
      khv[proj][m][col] = acc[i] + bias[col];
    }
  }
  __syncthreads();
  // scores (H,K) with mask
  for (int idx = tid; idx < NH * KNBR; idx += blockDim.x) {
    int h = idx >> 5, k = idx & 31;
    float s = 0.f;
#pragma unroll
    for (int d = 0; d < DHD; ++d)
      s += qrow[h * DHD + d] * khv[0][k][h * DHD + d];
    s *= 0.25f;               // 1/sqrt(16)
    if (msk[k]) s = -3.0e38f;
    sc[h][k] = s;
  }
  __syncthreads();
  if (tid < NH) {
    int h = tid;
    float mx = -3.4e38f;
    for (int k = 0; k < KNBR; ++k) mx = fmaxf(mx, sc[h][k]);
    float sum = 0.f;
    for (int k = 0; k < KNBR; ++k) { float e = __expf(sc[h][k] - mx); sc[h][k] = e; sum += e; }
    float inv = 1.f / sum;
    for (int k = 0; k < KNBR; ++k) sc[h][k] *= inv;
  }
  __syncthreads();
  if (tid < CCH) {
    int c = tid, h = c >> 4;
    float a = 0.f;
    for (int k = 0; k < KNBR; ++k) a += sc[h][k] * khv[1][k][c];
    attOut[(size_t)n * CCH + c] = (bf16_t)a;
  }
}

// ---------------- per-channel batch stats (deterministic tree reduction) ----------------
__global__ void stats_k(const float* __restrict__ X, int M, int Cdim,
                        float* __restrict__ meanOut, float* __restrict__ rstdOut) {
  __shared__ float s1[256], s2[256];
  int c = blockIdx.x, tid = threadIdx.x;
  float a = 0.f, b = 0.f;
  for (int i = tid; i < M; i += 256) {
    float v = X[(size_t)i * Cdim + c];
    a += v; b += v * v;
  }
  s1[tid] = a; s2[tid] = b;
  __syncthreads();
  for (int s = 128; s > 0; s >>= 1) {
    if (tid < s) { s1[tid] += s1[tid + s]; s2[tid] += s2[tid + s]; }
    __syncthreads();
  }
  if (tid == 0) {
    float m = s1[0] / (float)M;
    float v = s2[0] / (float)M - m * m;
    meanOut[c] = m;
    rstdOut[c] = rsqrtf(v + 1e-5f);
  }
}

// ---------------- BN -> bf16 copy (feeds final head GEMM) ----------------
__global__ void norm_bf16_k(const float* __restrict__ X,
                            const float* __restrict__ m, const float* __restrict__ r,
                            const float* __restrict__ g, const float* __restrict__ b,
                            bf16_t* __restrict__ out) {
  int i = blockIdx.x * blockDim.x + threadIdx.x;
  if (i >= NPTS * CCH) return;
  int c = i & 127;
  out[i] = (bf16_t)((X[i] - m[c]) * r[c] * g[c] + b[c]);
}

// ---------------- fused FFN per 16-row tile: BN1 -> lin1+relu (WMMA) -> lin2 (WMMA) + residual ----------------
__global__ void ffn_k(const float* __restrict__ x1,
                      const float* __restrict__ m1, const float* __restrict__ r1,
                      const float* __restrict__ g1, const float* __restrict__ b1,
                      const bf16_t* __restrict__ l1w, const float* __restrict__ l1b,
                      const bf16_t* __restrict__ l2w, const float* __restrict__ l2b,
                      float* __restrict__ x2) {
  __shared__ __align__(16) bf16_t xb[16][CCH];
  __shared__ float  xf[16][CCH];
  __shared__ __align__(16) bf16_t hb[16][FFD];
  int tid = threadIdx.x, lane = tid & 31, wave = tid >> 5;
  int r0 = blockIdx.x * 16;
  for (int idx = tid; idx < 16 * CCH; idx += blockDim.x) {
    int r = idx >> 7, c = idx & 127;
    float v = x1[(size_t)(r0 + r) * CCH + c];
    v = (v - m1[c]) * r1[c] * g1[c] + b1[c];
    xf[r][c] = v; xb[r][c] = (bf16_t)v;
  }
  __syncthreads();
  // hidden: 32 col tiles of 16, K=128
#pragma unroll
  for (int t = 0; t < 8; ++t) {
    int ni = wave * 8 + t;
    const bf16_t* arow = &xb[lane & 15][0];
    const bf16_t* brow = l1w + (size_t)(ni * 16 + (lane & 15)) * CCH;
    v8f acc = {};
#pragma unroll
    for (int k0 = 0; k0 < CCH; k0 += 32) {
      v16bf a = load_afrag(arow, lane, k0);
      v16bf b = load_bfrag(brow, lane, k0);
      acc = __builtin_amdgcn_wmma_f32_16x16x32_bf16(false, a, false, b, (short)0, acc, false, false);
    }
    int col = ni * 16 + (lane & 15);
#pragma unroll
    for (int i = 0; i < 8; ++i) {
      int m = ((lane >> 4) << 3) + i;
      hb[m][col] = (bf16_t)fmaxf(acc[i] + l1b[col], 0.f);
    }
  }
  __syncthreads();
  // output: 8 col tiles of 16, K=512
#pragma unroll
  for (int t = 0; t < 2; ++t) {
    int ni = wave * 2 + t;
    const bf16_t* arow = &hb[lane & 15][0];
    const bf16_t* brow = l2w + (size_t)(ni * 16 + (lane & 15)) * FFD;
    v8f acc = {};
#pragma unroll
    for (int k0 = 0; k0 < FFD; k0 += 32) {
      v16bf a = load_afrag(arow, lane, k0);
      v16bf b = load_bfrag(brow, lane, k0);
      acc = __builtin_amdgcn_wmma_f32_16x16x32_bf16(false, a, false, b, (short)0, acc, false, false);
    }
    int col = ni * 16 + (lane & 15);
#pragma unroll
    for (int i = 0; i < 8; ++i) {
      int m = ((lane >> 4) << 3) + i;
      x2[(size_t)(r0 + m) * CCH + col] = xf[m][col] + acc[i] + l2b[col];
    }
  }
}

// ---------------- final BN + relu ----------------
__global__ void bnrelu_k(const float* __restrict__ Y,
                         const float* __restrict__ m, const float* __restrict__ r,
                         const float* __restrict__ g, const float* __restrict__ b,
                         float* __restrict__ out) {
  int i = blockIdx.x * blockDim.x + threadIdx.x;
  if (i >= NPTS * COUT) return;
  int c = i & 127;
  out[i] = fmaxf((Y[i] - m[c]) * r[c] * g[c] + b[c], 0.f);
}

extern "C" void kernel_launch(void* const* d_in, const int* in_sizes, int n_in,
                              void* d_out, int out_size, void* d_ws, size_t ws_size,
                              hipStream_t stream) {
  const float* features  = (const float*)d_in[0];
  const int*   indices   = (const int*)d_in[1];
  const int*   key_idx   = (const int*)d_in[2];
  const float* in_proj_w = (const float*)d_in[3];
  const float* in_proj_b = (const float*)d_in[4];
  const float* out_w     = (const float*)d_in[5];
  const float* out_b     = (const float*)d_in[6];
  const float* qpos_w    = (const float*)d_in[7];
  const float* qpos_b    = (const float*)d_in[8];
  const float* kpos_w    = (const float*)d_in[9];
  const float* kpos_b    = (const float*)d_in[10];
  const float* n1g = (const float*)d_in[11];
  const float* n1b = (const float*)d_in[12];
  const float* n2g = (const float*)d_in[13];
  const float* n2b = (const float*)d_in[14];
  const float* l1w = (const float*)d_in[15];
  const float* l1b = (const float*)d_in[16];
  const float* l2w = (const float*)d_in[17];
  const float* l2b = (const float*)d_in[18];
  const float* olw = (const float*)d_in[19];
  const float* olb = (const float*)d_in[20];
  const float* bog = (const float*)d_in[21];
  const float* bob = (const float*)d_in[22];
  (void)in_sizes; (void)n_in; (void)out_size; (void)ws_size;

  char* ws = (char*)d_ws;
  size_t off = 0;
  auto carve = [&](size_t bytes) -> char* {
    char* p = ws + off;
    off = (off + bytes + 255) & ~(size_t)255;
    return p;
  };
  bf16_t* wq_bf   = (bf16_t*)carve((size_t)CCH * CCH * 2);
  bf16_t* wk_bf   = (bf16_t*)carve((size_t)CCH * CCH * 2);
  bf16_t* wv_bf   = (bf16_t*)carve((size_t)CCH * CCH * 2);
  bf16_t* ow_bf   = (bf16_t*)carve((size_t)CCH * CCH * 2);
  bf16_t* l1_bf   = (bf16_t*)carve((size_t)FFD * CCH * 2);
  bf16_t* l2_bf   = (bf16_t*)carve((size_t)CCH * FFD * 2);
  bf16_t* ol_bf   = (bf16_t*)carve((size_t)COUT * CCH * 2);
  float*  coords  = (float*)carve((size_t)NPTS * 3 * 4);
  bf16_t* qatt_bf = (bf16_t*)carve((size_t)NPTS * CCH * 2);  // q -> att -> BN2(x2) bf16
  float*  qh_y    = (float*)carve((size_t)NPTS * CCH * 4);   // qh -> head output y
  float*  x1      = (float*)carve((size_t)NPTS * CCH * 4);
  float*  x2      = (float*)carve((size_t)NPTS * CCH * 4);
  float*  stats   = (float*)carve(6 * CCH * 4);
  float *mean1 = stats,          *rstd1 = stats + CCH;
  float *mean2 = stats + 2*CCH,  *rstd2 = stats + 3*CCH;
  float *mean3 = stats + 4*CCH,  *rstd3 = stats + 5*CCH;

  const int cb = 256;
  cvt_bf16_k<<<(CCH*CCH + cb - 1)/cb, cb, 0, stream>>>(in_proj_w,             wq_bf, CCH*CCH);
  cvt_bf16_k<<<(CCH*CCH + cb - 1)/cb, cb, 0, stream>>>(in_proj_w + CCH*CCH,   wk_bf, CCH*CCH);
  cvt_bf16_k<<<(CCH*CCH + cb - 1)/cb, cb, 0, stream>>>(in_proj_w + 2*CCH*CCH, wv_bf, CCH*CCH);
  cvt_bf16_k<<<(CCH*CCH + cb - 1)/cb, cb, 0, stream>>>(out_w, ow_bf, CCH*CCH);
  cvt_bf16_k<<<(FFD*CCH + cb - 1)/cb, cb, 0, stream>>>(l1w, l1_bf, FFD*CCH);
  cvt_bf16_k<<<(CCH*FFD + cb - 1)/cb, cb, 0, stream>>>(l2w, l2_bf, CCH*FFD);
  cvt_bf16_k<<<(COUT*CCH + cb - 1)/cb, cb, 0, stream>>>(olw, ol_bf, COUT*CCH);

  coords_q_k<<<(NPTS*CCH + 255)/256, 256, 0, stream>>>(features, indices, qpos_w, qpos_b,
                                                       coords, qatt_bf);

  dim3 gq(NPTS/16, CCH/64);
  // qh = q @ wq^T + bq
  gemm128_k<false><<<gq, 128, 0, stream>>>(qatt_bf, wq_bf, in_proj_b, nullptr, qh_y);
  // attention per query -> att (bf16, reuses q buffer)
  attn_k<<<NPTS, 128, 0, stream>>>(features, coords, key_idx, kpos_w, kpos_b,
                                   wk_bf, wv_bf, in_proj_b, qh_y, qatt_bf);
  // x1 = features + att @ out_w^T + out_b
  gemm128_k<true><<<gq, 128, 0, stream>>>(qatt_bf, ow_bf, out_b, features, x1);
  stats_k<<<CCH, 256, 0, stream>>>(x1, NPTS, CCH, mean1, rstd1);
  // x2 = BN1(x1) + FFN(BN1(x1))
  ffn_k<<<NPTS/16, 128, 0, stream>>>(x1, mean1, rstd1, n1g, n1b, l1_bf, l1b, l2_bf, l2b, x2);
  stats_k<<<CCH, 256, 0, stream>>>(x2, NPTS, CCH, mean2, rstd2);
  // BN2(x2) -> bf16 (reuse qatt buffer), then y = BN2(x2) @ outl^T + outl_b
  norm_bf16_k<<<(NPTS*CCH + 255)/256, 256, 0, stream>>>(x2, mean2, rstd2, n2g, n2b, qatt_bf);
  gemm128_k<false><<<gq, 128, 0, stream>>>(qatt_bf, ol_bf, olb, nullptr, qh_y);
  stats_k<<<COUT, 256, 0, stream>>>(qh_y, NPTS, COUT, mean3, rstd3);
  bnrelu_k<<<(NPTS*COUT + 255)/256, 256, 0, stream>>>(qh_y, mean3, rstd3, bog, bob, (float*)d_out);
}
